// WLSMLPLayer_49065706389959
// MI455X (gfx1250) — matrix operand order repaired and checked by reading, not compiled, not run on hardware
//
#include <hip/hip_runtime.h>

typedef __bf16        v16bf __attribute__((ext_vector_type(16)));
typedef float         v8f   __attribute__((ext_vector_type(8)));
typedef unsigned int  u32x4 __attribute__((ext_vector_type(4)));
typedef unsigned int  u32x8 __attribute__((ext_vector_type(8)));
typedef float         f32x4 __attribute__((ext_vector_type(4)));

// MLP tile config: 64 rows per workgroup, 4 waves (wave32), each wave owns 16 rows.
#define TILE_M   64
#define NTHREADS 128
// LDS strides (elements), padded by 8 bf16 (16B) so rows stay 16B-aligned and
// fragment loads spread across banks.
#define XS_LD  136   // 128 + 8
#define H1_LD  264   // 256 + 8
#define H2_LD  520   // 512 + 8

// LDS element offsets (bf16 units)
#define XS_OFF  0
#define H1_OFF  (XS_OFF + TILE_M * XS_LD)     //  8704
#define H2_OFF  (H1_OFF + TILE_M * H1_LD)     // 25600
#define LDS_ELEMS (H2_OFF + TILE_M * H2_LD)   // 58880 elems = 117760 bytes

// Workspace layout (bf16 elements): weights pre-transposed to [N][K], K-contiguous.
#define W1T_OFF 0                       // 256 x 128
#define W2T_OFF (W1T_OFF + 256 * 128)   // 512 x 256
#define W3T_OFF (W2T_OFF + 512 * 256)   // 128 x 512

// ---- fragment loaders (layouts per CDNA5 ISA 7.12.2) ----
// A (16x32 bf16): lane holds row M=lane%16; elems 0..7 = K(h*8..h*8+7),
// elems 8..15 = K(16+h*8 .. 16+h*8+7).  (A in LDS)
__device__ __forceinline__ v16bf ld_a(const __bf16* rowp, int half) {
  const u32x4* p = (const u32x4*)(rowp + half * 8);
  u32x4 lo = p[0];        // elems 0..7
  u32x4 hi = p[2];        // +16 bf16 = +32B -> elems 8..15
  u32x8 w  = __builtin_shufflevector(lo, hi, 0, 1, 2, 3, 4, 5, 6, 7);
  return __builtin_bit_cast(v16bf, w);
}
// B (32x16 bf16): lane holds col N=lane%16; elems 0..7 = K(h*8..h*8+7),
// elems 8..15 = K(16+h*8 .. 16+h*8+7). Loaded straight from the pre-transposed global weight
// matrix Wt[n][k] (K-contiguous rows) -> two 16B loads, L0/L2 resident.
__device__ __forceinline__ v16bf ld_b(const __bf16* colp, int half) {
  const u32x4* p = (const u32x4*)(colp + half * 8);
  u32x4 lo = p[0];
  u32x4 hi = p[2];
  u32x8 w  = __builtin_shufflevector(lo, hi, 0, 1, 2, 3, 4, 5, 6, 7);
  return __builtin_bit_cast(v16bf, w);
}

// One tile-level GEMM: C[64 x N] = act(A[64 x K] @ W + bias), W given as
// bf16 Wt[N][K]. A is bf16 in LDS (stride lda). No barriers inside the
// k-loop: B comes from global (cache-resident), A from LDS written in a
// previous stage (guarded by the single barrier at function entry).
__device__ __forceinline__ void tile_gemm(
    const __bf16* __restrict__ A, int lda, int K,
    const __bf16* __restrict__ Wt, const float* __restrict__ bias, int N,
    bool do_relu,
    __bf16* __restrict__ Hout, int ldh,
    float* __restrict__ Gout, long long row0, int n_nodes, int tid) {
  const int lane = tid & 31;
  const int wave = tid >> 5;       // 0..3 -> rows [wave*16, wave*16+16)
  const int lrow = lane & 15;
  const int half = lane >> 4;

  __syncthreads();  // A tile (Xs/H1/H2) fully written by all waves

  const __bf16* arow = A + (wave * 16 + lrow) * lda;

  for (int n0 = 0; n0 < N; n0 += 64) {
    v8f acc[4];
#pragma unroll
    for (int f = 0; f < 4; ++f) {
      float b = bias[n0 + f * 16 + lrow];
#pragma unroll
      for (int r = 0; r < 8; ++r) acc[f][r] = b;
    }

    const __bf16* bcol0 = Wt + (long long)(n0 + lrow) * K;
    for (int k0 = 0; k0 < K; k0 += 32) {
      v16bf afrag = ld_a(arow + k0, half);
#pragma unroll
      for (int f = 0; f < 4; ++f) {
        v16bf bfrag = ld_b(bcol0 + (long long)(f * 16) * K + k0, half);
        acc[f] = __builtin_amdgcn_wmma_f32_16x16x32_bf16(
            false, afrag, false, bfrag, (short)0, acc[f], false, false);
      }
    }

    // Epilogue. C layout: VGPR r -> M = r + 8*half, N = lane%16.
#pragma unroll
    for (int f = 0; f < 4; ++f) {
#pragma unroll
      for (int r = 0; r < 8; ++r) {
        float v = acc[f][r];
        if (do_relu) v = v > 0.0f ? v : 0.0f;
        int mloc = wave * 16 + r + 8 * half;
        int col  = n0 + f * 16 + lrow;
        if (Hout) {
          Hout[mloc * ldh + col] = (__bf16)v;
        } else {
          long long rg = row0 + mloc;
          if (rg < (long long)n_nodes) Gout[rg * 256 + col] = v;
        }
      }
    }
  }
}

// Fused per-node MLP: 128 -> 256 -> 512 -> 128, intermediates in LDS (bf16).
// Writes h (fp32) into out[:, 0:128] (row stride 256).
__global__ void __launch_bounds__(NTHREADS)
mlp_kernel(const float* __restrict__ feats,
           const __bf16* __restrict__ W1t, const float* __restrict__ b1,
           const __bf16* __restrict__ W2t, const float* __restrict__ b2,
           const __bf16* __restrict__ W3t, const float* __restrict__ b3,
           float* __restrict__ out, int n_nodes) {
  extern __shared__ __align__(16) char smem_raw[];
  __bf16* lds = (__bf16*)smem_raw;
  __bf16* Xs = lds + XS_OFF;
  __bf16* H1 = lds + H1_OFF;
  __bf16* H2 = lds + H2_OFF;

  const int tid = threadIdx.x;
  const long long m0 = (long long)blockIdx.x * TILE_M;

  // Stage X tile: fp32 -> bf16, zero-pad out-of-range rows.
  for (int q = tid; q < TILE_M * 32; q += NTHREADS) {
    int row = q >> 5;        // 0..63
    int c4  = q & 31;        // float4 column
    f32x4 v = {0.f, 0.f, 0.f, 0.f};
    if (m0 + row < (long long)n_nodes)
      v = *(const f32x4*)(feats + (m0 + row) * 128 + c4 * 4);
    __bf16* p = Xs + row * XS_LD + c4 * 4;
    p[0] = (__bf16)v[0]; p[1] = (__bf16)v[1];
    p[2] = (__bf16)v[2]; p[3] = (__bf16)v[3];
  }

  tile_gemm(Xs, XS_LD, 128, W1t, b1, 256, true,  H1, H1_LD, nullptr, 0, 0, tid);
  tile_gemm(H1, H1_LD, 256, W2t, b2, 512, true,  H2, H2_LD, nullptr, 0, 0, tid);
  tile_gemm(H2, H2_LD, 512, W3t, b3, 128, false, nullptr, 0, out, m0, n_nodes, tid);
}

// Pre-convert + pre-transpose one weight matrix: W[K][N] fp32 -> Wt[N][K] bf16.
// Reads coalesced; one-shot (229K elements total across the three layers).
__global__ void transpose_w_kernel(const float* __restrict__ W,
                                   __bf16* __restrict__ Wt, int K, int N) {
  int t = blockIdx.x * 256 + threadIdx.x;
  if (t >= K * N) return;
  int n = t % N;
  int k = t / N;
  Wt[(long long)n * K + k] = (__bf16)W[t];
}

// Zero the aggregation half out[:, 128:256].
__global__ void zero_agg_kernel(float* __restrict__ out, long long n_nodes) {
  long long i = (long long)blockIdx.x * 256 + threadIdx.x;
  long long total = n_nodes * 128;
  if (i < total) {
    long long row = i >> 7;
    int col = (int)(i & 127);
    out[row * 256 + 128 + col] = 0.0f;
  }
}

// One wave per edge: lane loads 16B of h[src] (= out[src, 0:128], L2 resident)
// and does 4 fp32 atomic adds into out[dst, 128:256].
__global__ void __launch_bounds__(256)
edge_agg_kernel(const int* __restrict__ src, const int* __restrict__ dst,
                float* __restrict__ out, int n_edges) {
  int e = blockIdx.x * 8 + (threadIdx.x >> 5);
  if (e >= n_edges) return;
  int lane = threadIdx.x & 31;
  long long s = (long long)src[e];
  long long d = (long long)dst[e];
  f32x4 v = *(const f32x4*)(out + s * 256 + lane * 4);
  float* p = out + d * 256 + 128 + lane * 4;
#pragma unroll
  for (int i = 0; i < 4; ++i)
    (void)__hip_atomic_fetch_add(p + i, v[i], __ATOMIC_RELAXED,
                                 __HIP_MEMORY_SCOPE_AGENT);
}

extern "C" void kernel_launch(void* const* d_in, const int* in_sizes, int n_in,
                              void* d_out, int out_size, void* d_ws, size_t ws_size,
                              hipStream_t stream) {
  const float* feats = (const float*)d_in[0];
  const int*   src   = (const int*)d_in[1];
  const int*   dst   = (const int*)d_in[2];
  const float* W1    = (const float*)d_in[3];
  const float* b1    = (const float*)d_in[4];
  const float* W2    = (const float*)d_in[5];
  const float* b2    = (const float*)d_in[6];
  const float* W3    = (const float*)d_in[7];
  const float* b3    = (const float*)d_in[8];
  float* out = (float*)d_out;

  __bf16* wsb = (__bf16*)d_ws;
  __bf16* W1t = wsb + W1T_OFF;
  __bf16* W2t = wsb + W2T_OFF;
  __bf16* W3t = wsb + W3T_OFF;

  const int n_nodes = in_sizes[0] / 128;
  const int n_edges = in_sizes[1];

  // One-shot weight convert+transpose into workspace (bf16, [N][K]).
  transpose_w_kernel<<<(128 * 256 + 255) / 256, 256, 0, stream>>>(W1, W1t, 128, 256);
  transpose_w_kernel<<<(256 * 512 + 255) / 256, 256, 0, stream>>>(W2, W2t, 256, 512);
  transpose_w_kernel<<<(512 * 128 + 255) / 256, 256, 0, stream>>>(W3, W3t, 512, 128);

  const size_t smem = (size_t)LDS_ELEMS * sizeof(unsigned short);  // 117760 B
  const int mblocks = (n_nodes + TILE_M - 1) / TILE_M;
  mlp_kernel<<<mblocks, NTHREADS, smem, stream>>>(feats, W1t, b1, W2t, b2,
                                                  W3t, b3, out, n_nodes);

  long long ztotal = (long long)n_nodes * 128;
  int zblocks = (int)((ztotal + 255) / 256);
  zero_agg_kernel<<<zblocks, 256, 0, stream>>>(out, (long long)n_nodes);

  int eblocks = (n_edges + 7) / 8;
  edge_agg_kernel<<<eblocks, 256, 0, stream>>>(src, dst, out, n_edges);
}
